// FullAttention_37014028157212
// MI455X (gfx1250) — matrix operand order repaired and checked by reading, not compiled, not run on hardware
//
#include <hip/hip_runtime.h>
#include <cstdint>

typedef __attribute__((ext_vector_type(16))) _Float16 v16h;
typedef __attribute__((ext_vector_type(8)))  _Float16 v8h;
typedef __attribute__((ext_vector_type(8)))  float    v8f;
typedef __attribute__((ext_vector_type(4)))  float    v4f;
typedef __attribute__((__vector_size__(16))) int      v4i;

#define B_  2
#define L_  2048
#define S_  2048
#define H_  16
#define E_  64
#define D_  64
#define BM  64      // Q rows per block (4 waves x 16)
#define BN  64      // S columns per iteration (4 x 16-col fragments)
#define NW  4       // waves per block

// ---- async global->LDS path (CDNA5), with safe fallback ----
#if defined(__has_builtin)
#  if __has_builtin(__builtin_amdgcn_global_load_async_to_lds_b128)
#    define USE_ASYNC 1
#  endif
#endif

#ifdef USE_ASYNC
typedef v4i __attribute__((address_space(1))) ga_v4i;
typedef v4i __attribute__((address_space(3))) ls_v4i;
#endif

__device__ __forceinline__ void copy16_async(void* lds, const void* g) {
#ifdef USE_ASYNC
  __builtin_amdgcn_global_load_async_to_lds_b128(
      (ga_v4i*)(uintptr_t)g,
      (ls_v4i*)(uint32_t)(uintptr_t)lds,
      0, 0);
#else
  *(v4i*)lds = *(const v4i*)g;
#endif
}

__device__ __forceinline__ void wait_async_all() {
#ifdef USE_ASYNC
  asm volatile("s_wait_asynccnt 0x0" ::: "memory");
#endif
}

__device__ __forceinline__ float hmax16(float v) {
  v = fmaxf(v, __shfl_xor(v, 1, 32));
  v = fmaxf(v, __shfl_xor(v, 2, 32));
  v = fmaxf(v, __shfl_xor(v, 4, 32));
  v = fmaxf(v, __shfl_xor(v, 8, 32));
  return v;
}
__device__ __forceinline__ float hsum16(float v) {
  v += __shfl_xor(v, 1, 32);
  v += __shfl_xor(v, 2, 32);
  v += __shfl_xor(v, 4, 32);
  v += __shfl_xor(v, 8, 32);
  return v;
}

// =====================================================================
// Prep kernel: fp32 -> f16 repack.
//   Qh, Kh : [B][H][X][E]  (head-major, X = L or S)
//   Vh     : [B][H][D][S]  (transposed: d-major rows, contiguous in s)
// =====================================================================
__global__ __launch_bounds__(256)
void cvt_pack_kernel(const float* __restrict__ Q, const float* __restrict__ K,
                     const float* __restrict__ V,
                     _Float16* __restrict__ Qh, _Float16* __restrict__ Kh,
                     _Float16* __restrict__ Vh) {
  const int which = blockIdx.y;                   // 0=Q, 1=K, 2=V
  const size_t flat = ((size_t)blockIdx.x * 256 + threadIdx.x) * 8;

  if (which < 2) {
    // out index = flat over [B][H][X][E]; 8 contiguous e
    int    e  = (int)(flat & 63);
    size_t r  = flat >> 6;               // (b*H + h)*2048 + x
    int    x  = (int)(r & (L_ - 1));
    size_t bh = r >> 11;
    int    h  = (int)(bh & (H_ - 1));
    int    b  = (int)(bh >> 4);
    const float* src = (which == 0 ? Q : K) + (((size_t)b * L_ + x) * H_ + h) * E_ + e;
    v4f a = *(const v4f*)src;
    v4f c = *(const v4f*)(src + 4);
    v8h o;
#pragma unroll
    for (int i = 0; i < 4; ++i) { o[i] = (_Float16)a[i]; o[4 + i] = (_Float16)c[i]; }
    *(v8h*)((which == 0 ? Qh : Kh) + flat) = o;
  } else {
    // out index = flat over [B][H][D][S]; 8 contiguous s (gather strided in)
    int    s  = (int)(flat & (S_ - 1));
    size_t r  = flat >> 11;              // (b*H + h)*64 + d
    int    d  = (int)(r & (D_ - 1));
    size_t bh = r >> 6;
    int    h  = (int)(bh & (H_ - 1));
    int    b  = (int)(bh >> 4);
    const float* src = V + (((size_t)b * S_ + s) * H_ + h) * D_ + d;
    v8h o;
#pragma unroll
    for (int i = 0; i < 8; ++i) o[i] = (_Float16)src[(size_t)i * (H_ * D_)];
    *(v8h*)(Vh + flat) = o;
  }
}

// stage f16 K tile (64 s-rows x 64 e) and transposed V tile (64 d x 64 s):
// each tile = 8KB = 512 x 16B chunks; 128 threads x 4 chunks per tensor
__device__ __forceinline__ void stage_tiles(_Float16* kb, _Float16* vb,
                                            const _Float16* Kh, const _Float16* Vh,
                                            size_t khbase, size_t vhbase,
                                            int s0, int tid) {
#pragma unroll
  for (int i = 0; i < 4; ++i) {
    int c = tid + i * 128;                      // 0..511
    // K: row = c>>3 (8 chunks per 64-half row), col = (c&7)*8
    copy16_async(kb + (c << 3),
                 Kh + khbase + (size_t)(s0 + (c >> 3)) * E_ + ((c & 7) << 3));
    // V^T: d-row = c>>3, scol = (c&7)*8
    copy16_async(vb + (c << 3),
                 Vh + vhbase + (size_t)(c >> 3) * S_ + s0 + ((c & 7) << 3));
  }
}

__global__ __launch_bounds__(128)
void fa_fwd_kernel(const _Float16* __restrict__ Qh, const _Float16* __restrict__ Kh,
                   const _Float16* __restrict__ Vh, const float* __restrict__ Mp,
                   float* __restrict__ O) {
  __shared__ __attribute__((aligned(32))) _Float16 kbuf[2][BN * E_];  // 16 KB
  __shared__ __attribute__((aligned(32))) _Float16 vbuf[2][D_ * BN];  // 16 KB (transposed)
  __shared__ __attribute__((aligned(32))) _Float16 pshm[NW][16 * BN]; // 8 KB

  const int tid  = threadIdx.x;
  const int lane = tid & 31;
  const int wid  = tid >> 5;
  const int bh   = blockIdx.y;
  const int b    = bh / H_;
  const int h    = bh % H_;
  const int lw   = blockIdx.x * BM + wid * 16;  // wave's first Q row

  const int half = (lane >> 4) & 1;  // 0: lanes 0-15, 1: lanes 16-31
  const int ln   = lane & 15;

  // ---- load Q as two 16x32 f16 A-fragments (E = 64) ----
  const _Float16* qrow = Qh + (((size_t)b * H_ + h) * L_ + (lw + ln)) * E_;
  v16h qf[2];
#pragma unroll
  for (int e0 = 0; e0 < 2; ++e0) {
    // A-layout: j=0..15 -> k = half*16 + j
    v8h q0 = *(const v8h*)(qrow + e0 * 32 + (half << 4));
    v8h q1 = *(const v8h*)(qrow + e0 * 32 + 8 + (half << 4));
#pragma unroll
    for (int j = 0; j < 8; ++j) { qf[e0][j] = q0[j]; qf[e0][j + 8] = q1[j]; }
  }

  const v8f vzero = {};
  v8f acc[4];
#pragma unroll
  for (int d = 0; d < 4; ++d) acc[d] = vzero;
  float m_i[8], l_i[8];
#pragma unroll
  for (int r = 0; r < 8; ++r) { m_i[r] = -1e30f; l_i[r] = 0.0f; }

  const float scale = 0.125f;  // 1/sqrt(64)
  const size_t khbase = ((size_t)b * H_ + h) * (size_t)S_ * E_;
  const size_t vhbase = ((size_t)b * H_ + h) * (size_t)D_ * S_;

  stage_tiles(kbuf[0], vbuf[0], Kh, Vh, khbase, vhbase, 0, tid);

  const int T = S_ / BN;  // 32 tiles
  for (int t = 0; t < T; ++t) {
    wait_async_all();
    __syncthreads();  // tile t resident; all waves done with buffer (t+1)&1
    if (t + 1 < T)
      stage_tiles(kbuf[(t + 1) & 1], vbuf[(t + 1) & 1], Kh, Vh,
                  khbase, vhbase, (t + 1) * BN, tid);

    const _Float16* kt = kbuf[t & 1];
    const _Float16* vt = vbuf[t & 1];
    const int s0 = t * BN;

    // ---- scores = Q * K^T for 64 columns (four 16-col fragments) ----
    float p[4][8];
    float rowmax[8];
#pragma unroll
    for (int r = 0; r < 8; ++r) rowmax[r] = -1e30f;

#pragma unroll
    for (int f = 0; f < 4; ++f) {
      v8f c = vzero;
#pragma unroll
      for (int e0 = 0; e0 < 2; ++e0) {
        // B-frag: lane = column (s0+16f+ln); halves j -> kk = e0*32 + half*16 + j
        v16h bf = *(const v16h*)(kt + (f * 16 + ln) * E_ + e0 * 32 + (half << 4));
        c = __builtin_amdgcn_wmma_f32_16x16x32_f16(false, qf[e0], false, bf,
                                                   (short)0, c, false, false);
      }
#pragma unroll
      for (int r = 0; r < 8; ++r) {
        int mrow = r + (half << 3);
        float msk = Mp[(size_t)(lw + mrow) * S_ + (s0 + f * 16 + ln)];
        float tv  = (c[r] + msk) * scale;
        p[f][r]   = tv;
        rowmax[r] = fmaxf(rowmax[r], tv);
      }
    }

    // ---- online softmax update (one reduction pass per row per 64 cols) ----
#pragma unroll
    for (int r = 0; r < 8; ++r) {
      float mx   = hmax16(rowmax[r]);
      float mnew = fmaxf(m_i[r], mx);
      float corr = __expf(m_i[r] - mnew);
      m_i[r] = mnew;
      float s = 0.0f;
#pragma unroll
      for (int f = 0; f < 4; ++f) {
        float e = __expf(p[f][r] - mnew);
        p[f][r] = e;
        s += e;
      }
      l_i[r] = l_i[r] * corr + hsum16(s);
#pragma unroll
      for (int d = 0; d < 4; ++d) acc[d][r] *= corr;
    }

    // ---- transpose P (C-layout -> A-layout) through per-wave LDS ----
    _Float16* ps = &pshm[wid][0];
#pragma unroll
    for (int f = 0; f < 4; ++f)
#pragma unroll
      for (int r = 0; r < 8; ++r)
        ps[(r + (half << 3)) * BN + f * 16 + ln] = (_Float16)p[f][r];

    __builtin_amdgcn_wave_barrier();
    asm volatile("s_wait_dscnt 0x0" ::: "memory");
    __builtin_amdgcn_wave_barrier();

    // two 16x32 A-fragments of P (k-chunks over s-local 0..31, 32..63)
    const _Float16* prow = ps + ln * BN;
    v16h pf[2];
#pragma unroll
    for (int kc = 0; kc < 2; ++kc) {
      v8h p0 = *(const v8h*)(prow + kc * 32 + (half << 4));
      v8h p1 = *(const v8h*)(prow + kc * 32 + 8 + (half << 4));
#pragma unroll
      for (int j = 0; j < 8; ++j) { pf[kc][j] = p0[j]; pf[kc][j + 8] = p1[j]; }
    }

    // ---- O += P (16x64) * V (64x16) for four 16-wide D chunks ----
#pragma unroll
    for (int d = 0; d < 4; ++d) {
#pragma unroll
      for (int kc = 0; kc < 2; ++kc) {
        // V^T tile: row (d*16+ln) holds 64 s-halves; kk = kc*32 + half*16 + j
        v16h vf = *(const v16h*)(vt + (d * 16 + ln) * BN + kc * 32 + (half << 4));
        acc[d] = __builtin_amdgcn_wmma_f32_16x16x32_f16(false, pf[kc], false, vf,
                                                        (short)0, acc[d], false, false);
      }
    }

    __syncthreads();  // all waves finished reading buffer t&1
  }

  // ---- epilogue: O = acc / l ----
#pragma unroll
  for (int r = 0; r < 8; ++r) {
    int mrow  = r + (half << 3);
    float inv = 1.0f / l_i[r];
    size_t base = (((size_t)b * L_ + (lw + mrow)) * H_ + h) * D_;
#pragma unroll
    for (int d = 0; d < 4; ++d)
      O[base + d * 16 + ln] = acc[d][r] * inv;
  }
}

extern "C" void kernel_launch(void* const* d_in, const int* in_sizes, int n_in,
                              void* d_out, int out_size, void* d_ws, size_t ws_size,
                              hipStream_t stream) {
  (void)in_sizes; (void)n_in; (void)out_size; (void)ws_size;
  const float* Q = (const float*)d_in[0];
  const float* K = (const float*)d_in[1];
  const float* V = (const float*)d_in[2];
  const float* M = (const float*)d_in[3];
  float* O = (float*)d_out;

  const size_t N = (size_t)B_ * H_ * L_ * E_;   // 4,194,304 elements per tensor
  _Float16* Qh = (_Float16*)d_ws;
  _Float16* Kh = Qh + N;
  _Float16* Vh = Kh + N;

  // repack/convert: 3 tensors, 8 elements per thread
  dim3 cgrid((unsigned)(N / (8 * 256)), 3);
  cvt_pack_kernel<<<cgrid, dim3(256), 0, stream>>>(Q, K, V, Qh, Kh, Vh);

  dim3 grid(L_ / BM, B_ * H_);
  fa_fwd_kernel<<<grid, dim3(128), 0, stream>>>(Qh, Kh, Vh, M, O);
}
